// CrossModalAttention_10840497455134
// MI455X (gfx1250) — compile-verified, matrix operand order checked by reading, not run on hardware
//
#include <hip/hip_runtime.h>
#include <cstdint>

// ---------------------------------------------------------------- constants
#define B_   2
#define T_   2048
#define D_   1024
#define H_   16
#define HD_  64
#define BT_  (B_ * T_)          // 4096
#define EPS_ 1e-6f

typedef __bf16 bf16;
typedef __attribute__((ext_vector_type(16))) __bf16 v16bf;
typedef __attribute__((ext_vector_type(8)))  __bf16 v8bf;
typedef __attribute__((ext_vector_type(8)))  float  v8f;

// ---------------------------------------------------------------- WMMA core
__device__ __forceinline__ v8f wmma_bf16(v16bf a, v16bf b, v8f c) {
  // D = A(16x32) * B(32x16) + C(16x16 f32)
  return __builtin_amdgcn_wmma_f32_16x16x32_bf16(
      /*neg_a=*/false, a, /*neg_b=*/false, b,
      /*c_mod=*/(short)0, c, /*reuse_a=*/false, /*reuse_b=*/false);
}

// Load one A or B operand tile (16 rows/cols x K=32, bf16) in the wave32 WMMA
// register layout.  src is row-major [* x ld]; lanes 0-15 take K {0..7,16..23},
// lanes 16-31 take K {8..15,24..31} (ISA 7.12.2).  Works for global or LDS
// pointers (LDS reads lower to ds_load_b128).
__device__ __forceinline__ v16bf load_tile_bf16(const bf16* __restrict__ src,
                                                int ld, int row, int k0,
                                                int lane) {
  const int kb = k0 + ((lane >> 4) << 3);           // +0 or +8
  const bf16* p = src + (size_t)row * ld + kb;
  v8bf lo = *(const v8bf*)(p);                      // K kb..kb+7
  v8bf hi = *(const v8bf*)(p + 16);                 // K kb+16..kb+23
  v16bf out;
#pragma unroll
  for (int i = 0; i < 8; ++i) { out[i] = lo[i]; out[i + 8] = hi[i]; }
  return out;
}

// CDNA5 async copy: 16 bytes/lane from global memory straight into LDS,
// tracked with ASYNCcnt (no VGPR round trip).  lds_addr is the wave-relative
// LDS byte address (low 32 bits of the generic LDS pointer).
__device__ __forceinline__ void async_copy_b128(unsigned lds_addr,
                                                const bf16* gaddr) {
  asm volatile("global_load_async_to_lds_b128 %0, %1, off"
               :: "v"(lds_addr), "v"((uint64_t)(uintptr_t)gaddr)
               : "memory");
}
__device__ __forceinline__ void async_wait0() {
  asm volatile("s_wait_asynccnt 0x0" ::: "memory");
}
__device__ __forceinline__ unsigned lds_off(const void* p) {
  return (unsigned)(uintptr_t)p;                    // low 32 bits = LDS offset
}

// Butterfly reductions within each 16-lane half (rows of the C layout).
__device__ __forceinline__ float hred_max(float x) {
  x = fmaxf(x, __shfl_xor(x, 1, 32));
  x = fmaxf(x, __shfl_xor(x, 2, 32));
  x = fmaxf(x, __shfl_xor(x, 4, 32));
  x = fmaxf(x, __shfl_xor(x, 8, 32));
  return x;
}
__device__ __forceinline__ float hred_add(float x) {
  x += __shfl_xor(x, 1, 32);
  x += __shfl_xor(x, 2, 32);
  x += __shfl_xor(x, 4, 32);
  x += __shfl_xor(x, 8, 32);
  return x;
}

// ---------------------------------------------------------------- f32 -> bf16
__global__ __launch_bounds__(256) void cvt_kernel(const float* __restrict__ in,
                                                  bf16* __restrict__ out,
                                                  int n) {
  int i = blockIdx.x * blockDim.x + threadIdx.x;
  int stride = gridDim.x * blockDim.x;
  for (; i < n; i += stride) out[i] = (bf16)in[i];
}

// ---------------------------------------------------------------- GEMM + bias
// C[m, j] = sum_k X[m, k] * W[j, k] + bias[j]
// Block: 4 waves, each a 32x64 tile (8 WMMAs / K-step); the shared W tile
// (64 x 32 bf16, 4 KB) is staged once per block via async-to-LDS.
// mode 0: bf16 out [B,H,T,HD]   (Q / K)
// mode 1: bf16 out [B,H,HD,T]   (V, key-contiguous for PV's B operand)
// mode 2: f32  out [B*T, D]     (final output projection)
__global__ __launch_bounds__(128) void gemm_bias_kernel(
    const bf16* __restrict__ X, const bf16* __restrict__ W,
    const float* __restrict__ bias, void* __restrict__ outp, int mode) {
  __shared__ __attribute__((aligned(16))) bf16 wtile[64 * 32];
  const int tid = threadIdx.x;
  const int lane = tid & 31;
  const int wave = tid >> 5;
  const int m0 = (blockIdx.x * 4 + wave) * 32;      // 32-token row tile
  const int nbase = blockIdx.y * 64;                // one head-width of columns
  const int nloc = lane & 15;
  const unsigned wl = lds_off(wtile);

  // this thread's two 16B chunks of the 64x32 W tile (4 chunks per 64B row)
  const int wr0 = tid >> 2,          ws0 = tid & 3;
  const int wr1 = (tid + 128) >> 2,  ws1 = tid & 3;
  const unsigned wo0 = wl + (unsigned)(wr0 * 64 + ws0 * 16);
  const unsigned wo1 = wl + (unsigned)(wr1 * 64 + ws1 * 16);

  v8f acc[2][4] = {};
  for (int k0 = 0; k0 < D_; k0 += 32) {
    // stage W[nbase..+63][k0..+31] -> LDS (straight-line, no guards)
    async_copy_b128(wo0, W + (size_t)(nbase + wr0) * D_ + k0 + ws0 * 8);
    async_copy_b128(wo1, W + (size_t)(nbase + wr1) * D_ + k0 + ws1 * 8);
    async_wait0();
    __syncthreads();
    const v16bf a0 = load_tile_bf16(X, D_, m0 + nloc, k0, lane);
    const v16bf a1 = load_tile_bf16(X, D_, m0 + 16 + nloc, k0, lane);
#pragma unroll
    for (int jt = 0; jt < 4; ++jt) {
      const v16bf b = load_tile_bf16(wtile, 32, jt * 16 + nloc, 0, lane);
      acc[0][jt] = wmma_bf16(a0, b, acc[0][jt]);
      acc[1][jt] = wmma_bf16(a1, b, acc[1][jt]);
    }
    __syncthreads();                                // protect WAR on wtile
  }

  const int mh = (lane >> 4) << 3;                  // C layout: M = r or r+8
#pragma unroll
  for (int at = 0; at < 2; ++at) {
#pragma unroll
    for (int jt = 0; jt < 4; ++jt) {
      const int j = nbase + jt * 16 + nloc;
      const float bj = bias[j];
#pragma unroll
      for (int r = 0; r < 8; ++r) {
        const int m = m0 + at * 16 + mh + r;        // global token row
        const float val = acc[at][jt][r] + bj;
        const int b = m / T_, tt = m % T_;
        if (mode == 0) {
          const int h = j >> 6, hd = j & 63;
          ((bf16*)outp)[(((size_t)(b * H_ + h)) * T_ + tt) * HD_ + hd] =
              (bf16)val;
        } else if (mode == 1) {
          const int h = j >> 6, hd = j & 63;
          ((bf16*)outp)[(((size_t)(b * H_ + h)) * HD_ + hd) * T_ + tt] =
              (bf16)val;
        } else {
          ((float*)outp)[(size_t)m * D_ + j] = val;
        }
      }
    }
  }
}

// ---------------------------------------------------------------- RMSNorm
// In-place per-row (HD=64) rmsnorm on bf16; one wave per row (2 elems/lane).
// `post` additionally folds the attention scale (1/sqrt(HD)=0.125) into Q.
__global__ __launch_bounds__(256) void rmsnorm_kernel(bf16* __restrict__ x,
                                                      const float* __restrict__ w,
                                                      float post, int nrows) {
  const int lane = threadIdx.x & 31;
  const int wave = threadIdx.x >> 5;
  const int row = blockIdx.x * 8 + wave;
  if (row >= nrows) return;
  bf16* p = x + (size_t)row * HD_ + lane * 2;
  const float x0 = (float)p[0], x1 = (float)p[1];
  float ss = x0 * x0 + x1 * x1;
  ss += __shfl_xor(ss, 1, 32);
  ss += __shfl_xor(ss, 2, 32);
  ss += __shfl_xor(ss, 4, 32);
  ss += __shfl_xor(ss, 8, 32);
  ss += __shfl_xor(ss, 16, 32);
  const float r = rsqrtf(ss * (1.0f / HD_) + EPS_) * post;
  p[0] = (bf16)(x0 * r * w[lane * 2 + 0]);
  p[1] = (bf16)(x1 * r * w[lane * 2 + 1]);
}

// ---------------------------------------------------------------- attention
// Block: 4 waves (same b,h), each owning a 16-query tile.  Per 32-key chunk
// the block stages K (32x64) and V (64x32) once via async-to-LDS, then each
// wave runs S = Q K^T (4 wmma), online softmax, P restage (C->A layout via
// per-wave LDS), O += P V (4 wmma).
__global__ __launch_bounds__(128) void attn_kernel(const bf16* __restrict__ q,
                                                   const bf16* __restrict__ k,
                                                   const bf16* __restrict__ v,
                                                   bf16* __restrict__ o) {
  __shared__ __attribute__((aligned(16))) bf16 ktile[32 * 64];  // [key][d]
  __shared__ __attribute__((aligned(16))) bf16 vtile[64 * 32];  // [d][key]
  __shared__ __attribute__((aligned(16))) bf16 plds[4][16 * 32];
  const int tid = threadIdx.x;
  const int lane = tid & 31;
  const int wave = tid >> 5;
  const int bh = blockIdx.x;                        // 0..31
  const int q0 = (blockIdx.y * 4 + wave) * 16;
  const int nloc = lane & 15;
  const int mh = (lane >> 4) << 3;
  const unsigned kl = lds_off(ktile);
  const unsigned vl = lds_off(vtile);

  const bf16* qh = q + (size_t)bh * T_ * HD_;
  const bf16* kh = k + (size_t)bh * T_ * HD_;
  const bf16* vh = v + (size_t)bh * HD_ * T_;

  // this thread's staging chunks: K = 32 rows x 8 chunks, V = 64 rows x 4
  const int kr0 = tid >> 3,         ks0 = tid & 7;
  const int kr1 = (tid + 128) >> 3, ks1 = tid & 7;
  const unsigned ko0 = kl + (unsigned)(kr0 * 128 + ks0 * 16);
  const unsigned ko1 = kl + (unsigned)(kr1 * 128 + ks1 * 16);
  const int vr0 = tid >> 2,         vs0 = tid & 3;
  const int vr1 = (tid + 128) >> 2, vs1 = tid & 3;
  const unsigned vo0 = vl + (unsigned)(vr0 * 64 + vs0 * 16);
  const unsigned vo1 = vl + (unsigned)(vr1 * 64 + vs1 * 16);

  // Q tile (16 x 64) as two A operands; scale already folded in by rmsnorm.
  const v16bf qa0 = load_tile_bf16(qh, HD_, q0 + nloc, 0, lane);
  const v16bf qa1 = load_tile_bf16(qh, HD_, q0 + nloc, 32, lane);

  v8f O[4] = {};
  float mrow[8], lrow[8];
#pragma unroll
  for (int r = 0; r < 8; ++r) { mrow[r] = -1e30f; lrow[r] = 0.0f; }

  for (int kc = 0; kc < T_; kc += 32) {
    // ---- stage K/V tiles for the whole block (async -> LDS, no guards) --
    async_copy_b128(ko0, kh + (size_t)(kc + kr0) * HD_ + ks0 * 8);
    async_copy_b128(ko1, kh + (size_t)(kc + kr1) * HD_ + ks1 * 8);
    async_copy_b128(vo0, vh + (size_t)vr0 * T_ + kc + vs0 * 8);
    async_copy_b128(vo1, vh + (size_t)vr1 * T_ + kc + vs1 * 8);
    async_wait0();
    __syncthreads();

    // ---- S = Q * K^T for 32 keys (two 16-key N tiles) -------------------
    v8f s0 = {}, s1 = {};
    {
      const v16bf b00 = load_tile_bf16(ktile, 64, nloc, 0, lane);
      const v16bf b01 = load_tile_bf16(ktile, 64, nloc, 32, lane);
      s0 = wmma_bf16(qa0, b00, s0);
      s0 = wmma_bf16(qa1, b01, s0);
      const v16bf b10 = load_tile_bf16(ktile, 64, 16 + nloc, 0, lane);
      const v16bf b11 = load_tile_bf16(ktile, 64, 16 + nloc, 32, lane);
      s1 = wmma_bf16(qa0, b10, s1);
      s1 = wmma_bf16(qa1, b11, s1);
    }
    // ---- online softmax (per C-layout row r) ----------------------------
#pragma unroll
    for (int r = 0; r < 8; ++r) {
      const float cm = hred_max(fmaxf(s0[r], s1[r]));
      const float nm = fmaxf(mrow[r], cm);
      const float alpha = __expf(mrow[r] - nm);
      mrow[r] = nm;
      const float p0 = __expf(s0[r] - nm);
      const float p1 = __expf(s1[r] - nm);
      s0[r] = p0; s1[r] = p1;
      lrow[r] = lrow[r] * alpha + hred_add(p0 + p1);
#pragma unroll
      for (int t = 0; t < 4; ++t) O[t][r] *= alpha;
    }
    // ---- restage P (C layout -> A layout) through per-wave LDS ----------
    bf16* pl = plds[wave];
#pragma unroll
    for (int r = 0; r < 8; ++r) {
      pl[(mh + r) * 32 + nloc] = (bf16)s0[r];
      pl[(mh + r) * 32 + 16 + nloc] = (bf16)s1[r];
    }
    __syncthreads();
    const v16bf pa = load_tile_bf16(pl, 32, nloc, 0, lane);
    // ---- O += P * V ------------------------------------------------------
#pragma unroll
    for (int dt = 0; dt < 4; ++dt) {
      const v16bf vb = load_tile_bf16(vtile, 32, dt * 16 + nloc, 0, lane);
      O[dt] = wmma_bf16(pa, vb, O[dt]);
    }
    __syncthreads();                                // protect WAR on k/v tiles
  }

  // ---- epilogue: O / l  -> attn_out [B*T, D] bf16 -----------------------
  const int b = bh / H_, h = bh % H_;
#pragma unroll
  for (int dt = 0; dt < 4; ++dt) {
    const int d = h * HD_ + dt * 16 + nloc;
#pragma unroll
    for (int r = 0; r < 8; ++r) {
      const int t = q0 + mh + r;
      o[((size_t)(b * T_ + t)) * D_ + d] = (bf16)(O[dt][r] / lrow[r]);
    }
  }
}

// ---------------------------------------------------------------- launcher
extern "C" void kernel_launch(void* const* d_in, const int* in_sizes, int n_in,
                              void* d_out, int out_size, void* d_ws,
                              size_t ws_size, hipStream_t stream) {
  const float* query = (const float*)d_in[0];
  const float* key_value = (const float*)d_in[1];
  const float* Wq = (const float*)d_in[2];
  const float* bq = (const float*)d_in[3];
  const float* Wk = (const float*)d_in[4];
  const float* bk = (const float*)d_in[5];
  const float* Wv = (const float*)d_in[6];
  const float* bv = (const float*)d_in[7];
  const float* Wo = (const float*)d_in[8];
  const float* bo = (const float*)d_in[9];
  const float* qn_w = (const float*)d_in[10];
  const float* kn_w = (const float*)d_in[11];

  char* ws = (char*)d_ws;
  const size_t MB = 1u << 20;
  bf16* Xq  = (bf16*)(ws + 0 * MB);    // 4M elems (8 MB)
  bf16* Xkv = (bf16*)(ws + 8 * MB);    // 4M elems
  bf16* Wqb = (bf16*)(ws + 16 * MB);   // 1M elems (2 MB)
  bf16* Wkb = (bf16*)(ws + 18 * MB);
  bf16* Wvb = (bf16*)(ws + 20 * MB);
  bf16* Wob = (bf16*)(ws + 22 * MB);
  bf16* qb  = (bf16*)(ws + 24 * MB);   // [B,H,T,HD] bf16
  bf16* kb  = (bf16*)(ws + 32 * MB);   // [B,H,T,HD]
  bf16* vb  = (bf16*)(ws + 40 * MB);   // [B,H,HD,T]
  bf16* ab  = (bf16*)(ws + 48 * MB);   // [B*T, D] attention output

  const int nX = BT_ * D_;             // 4194304
  const int nW = D_ * D_;              // 1048576

  // 1. precision conversion (feed WMMA directly from bf16)
  cvt_kernel<<<dim3(2048), dim3(256), 0, stream>>>(query, Xq, nX);
  cvt_kernel<<<dim3(2048), dim3(256), 0, stream>>>(key_value, Xkv, nX);
  cvt_kernel<<<dim3(1024), dim3(256), 0, stream>>>(Wq, Wqb, nW);
  cvt_kernel<<<dim3(1024), dim3(256), 0, stream>>>(Wk, Wkb, nW);
  cvt_kernel<<<dim3(1024), dim3(256), 0, stream>>>(Wv, Wvb, nW);
  cvt_kernel<<<dim3(1024), dim3(256), 0, stream>>>(Wo, Wob, nW);

  // 2. Q/K/V projections (WMMA GEMM): grid = (BT/128, D/64), 32x64 per wave
  dim3 ggrid(BT_ / 128, D_ / 64), gblk(128);
  gemm_bias_kernel<<<ggrid, gblk, 0, stream>>>(Xq,  Wqb, bq, qb, 0);
  gemm_bias_kernel<<<ggrid, gblk, 0, stream>>>(Xkv, Wkb, bk, kb, 0);
  gemm_bias_kernel<<<ggrid, gblk, 0, stream>>>(Xkv, Wvb, bv, vb, 1);

  // 3. head rmsnorm (attention scale 0.125 folded into Q)
  const int nrows = B_ * H_ * T_;      // 65536
  rmsnorm_kernel<<<dim3(nrows / 8), dim3(256), 0, stream>>>(qb, qn_w, 0.125f, nrows);
  rmsnorm_kernel<<<dim3(nrows / 8), dim3(256), 0, stream>>>(kb, kn_w, 1.0f, nrows);

  // 4. flash attention: grid = (B*H, T/16/4)
  attn_kernel<<<dim3(B_ * H_, T_ / 64), dim3(128), 0, stream>>>(qb, kb, vb, ab);

  // 5. output projection -> f32 d_out
  gemm_bias_kernel<<<ggrid, gblk, 0, stream>>>(ab, Wob, bo, d_out, 2);
}
